// MaskedPositionsOnlyMLMObjective_69715909149483
// MI455X (gfx1250) — compile-verified, matrix operand order checked by reading, not run on hardware
//
#include <hip/hip_runtime.h>
#include <hip/hip_bf16.h>
#include <math.h>

// ---------------------------------------------------------------------------
// MaskedPositionsOnlyMLMObjective for MI455X (gfx1250, wave32)
//   bf16 WMMA streaming-softmax GEMM with async global->LDS double buffering
//   hidden (4096x768 f32), labels (4096 i32), lm_weight (32000x768 f32)
//   -> loss_sum, num_masked, num_correct  (3 floats in d_out)
// ---------------------------------------------------------------------------

typedef __attribute__((ext_vector_type(16))) __bf16 v16bf;
typedef __attribute__((ext_vector_type(8)))  float  v8f;

#define NTOK   4096
#define HDIM   768
#define VOCAB  32000
#define IGNORE (-100)

#define VCHUNK 1280                      // vocab columns per block (chunk)
#define NCHUNK (VOCAB / VCHUNK)          // 25 chunks
#define RPB    128                       // rows per block
#define NROWBLK (NTOK / RPB)             // 32 row blocks
#define VT     64                        // logit tile: RPB x VT
#define KT     32                        // k per WMMA step (bf16 K=32)
#define KSTEPS (HDIM / KT)               // 24

#define NEG_INF (-__builtin_inff())

union FragCast { uint4 u[2]; v16bf v; };

// Low 32 bits of a generic pointer into LDS == the LDS byte offset.
__device__ __forceinline__ unsigned lds_off(const void* p) {
    return (unsigned)(unsigned long long)p;
}

// CDNA5 async DMA: global -> LDS, 16B, tracked by ASYNCcnt (no VGPR round-trip)
__device__ __forceinline__ void async_copy16(unsigned dst_lds, const void* src) {
    asm volatile("global_load_async_to_lds_b128 %0, %1, off"
                 :: "v"(dst_lds), "v"(src) : "memory");
}

// -------------------- 1) fp32 -> bf16 conversion pass ----------------------
__global__ __launch_bounds__(256)
void convert_bf16_kernel(const float* __restrict__ W,
                         const float* __restrict__ Hst,
                         __bf16* __restrict__ Wb,
                         __bf16* __restrict__ Hb) {
    long i0 = ((long)blockIdx.x * 256 + threadIdx.x) * 4;
#pragma unroll
    for (int j = 0; j < 4; ++j) {
        long i = i0 + j;
        if (i < (long)VOCAB * HDIM) Wb[i] = (__bf16)W[i];
        if (i < (long)NTOK * HDIM)  Hb[i] = (__bf16)Hst[i];
    }
}

// -------------------- 2) streaming logits + online softmax ----------------
__global__ __launch_bounds__(256)
void mlm_logits_chunk_kernel(const __bf16* __restrict__ Hb,
                             const __bf16* __restrict__ Wb,
                             const int* __restrict__ labels,
                             float* __restrict__ pm, float* __restrict__ ps,
                             float* __restrict__ pt, float* __restrict__ pav,
                             int*   __restrict__ pai) {
    __shared__ __bf16 As[2][RPB][KT];   // 2 x 8 KB  hidden tile (rows x k)
    __shared__ __bf16 Bs[2][VT][KT];    // 2 x 4 KB  weight tile (cols x k)
    __shared__ float  Ls[RPB][VT];      // 32 KB logit tile
    __shared__ float  m_run[RPB], s_run[RPB], t_run[RPB], amax_v[RPB];
    __shared__ int    amax_i[RPB];

    const int tid   = threadIdx.x;
    const int wave  = tid >> 5;        // 0..7
    const int lane  = tid & 31;
    const int half  = lane >> 4;       // 0: lanes 0-15, 1: lanes 16-31
    const int l16   = lane & 15;
    const int chunk = blockIdx.x;                  // 0..NCHUNK-1
    const int rowBase = blockIdx.y * RPB;

    // per-thread staging coordinates (3 async 16B copies per thread per step)
    const int arow0 = tid >> 2,           aoff0 = (tid & 3) * 8;        // A chunk 0
    const int arow1 = (tid + 256) >> 2,   aoff1 = (tid & 3) * 8;        // A chunk 1
    const int brow  = tid >> 2,           boff  = (tid & 3) * 8;        // B chunk

    if (tid < RPB) {
        m_run[tid] = NEG_INF; s_run[tid] = 0.f; t_run[tid] = NEG_INF;
        amax_v[tid] = NEG_INF; amax_i[tid] = 0;
    }
    __syncthreads();

    for (int vt = 0; vt < VCHUNK / VT; ++vt) {
        const int vbase = chunk * VCHUNK + vt * VT;

        v8f acc[4];
#pragma unroll
        for (int j = 0; j < 4; ++j) acc[j] = (v8f){};

        // ---- prologue: async-stage k-step 0 into buffer 0 ----
        {
            const int k0 = 0;
            async_copy16(lds_off(&As[0][arow0][aoff0]),
                         &Hb[(long)(rowBase + arow0) * HDIM + k0 + aoff0]);
            async_copy16(lds_off(&As[0][arow1][aoff1]),
                         &Hb[(long)(rowBase + arow1) * HDIM + k0 + aoff1]);
            async_copy16(lds_off(&Bs[0][brow][boff]),
                         &Wb[(long)(vbase + brow) * HDIM + k0 + boff]);
        }

        for (int kt = 0; kt < KSTEPS; ++kt) {
            const int cur = kt & 1;

            // ---- issue async stage for k-step kt+1 into the other buffer ----
            if (kt + 1 < KSTEPS) {
                const int nb = cur ^ 1;
                const int k1 = (kt + 1) * KT;
                async_copy16(lds_off(&As[nb][arow0][aoff0]),
                             &Hb[(long)(rowBase + arow0) * HDIM + k1 + aoff0]);
                async_copy16(lds_off(&As[nb][arow1][aoff1]),
                             &Hb[(long)(rowBase + arow1) * HDIM + k1 + aoff1]);
                async_copy16(lds_off(&Bs[nb][brow][boff]),
                             &Wb[(long)(vbase + brow) * HDIM + k1 + boff]);
                // 3 in flight for kt+1; wait until kt's 3 have landed
                asm volatile("s_wait_asynccnt 0x3" ::: "memory");
            } else {
                asm volatile("s_wait_asynccnt 0x0" ::: "memory");
            }
            __syncthreads();   // kt's tiles visible to all waves

            // ---- A fragment (16x32 bf16, ISA lane layout) ----
            // lanes 0-15 : M=lane,    K = {0..7} then {16..23}
            // lanes 16-31: M=lane-16, K = {8..15} then {24..31}
            FragCast a;
            {
                const __bf16* base = &As[cur][wave * 16 + l16][half * 8];
                a.u[0] = *(const uint4*)(base);
                a.u[1] = *(const uint4*)(base + 16);
            }
            // ---- 4 WMMAs: 16x16 out each, covering 64 columns ----
#pragma unroll
            for (int j = 0; j < 4; ++j) {
                FragCast b;
                const __bf16* base = &Bs[cur][j * 16 + l16][half * 8];
                b.u[0] = *(const uint4*)(base);
                b.u[1] = *(const uint4*)(base + 16);
                acc[j] = __builtin_amdgcn_wmma_f32_16x16x32_bf16(
                    /*neg_a=*/false, a.v, /*neg_b=*/false, b.v,
                    /*c_mod=*/(short)0, acc[j],
                    /*reuse_a=*/false, /*reuse_b=*/false);
            }
            __syncthreads();   // all reads of buf `cur` done before it is re-filled
        }

        // ---- spill accumulators (C layout: lane half selects row+8) ----
#pragma unroll
        for (int j = 0; j < 4; ++j)
#pragma unroll
            for (int r = 0; r < 8; ++r)
                Ls[wave * 16 + half * 8 + r][j * 16 + l16] = acc[j][r];
        __syncthreads();

        // ---- per-row online softmax / argmax / target update ----
        if (tid < RPB) {
            const int r = tid;
            float tm = NEG_INF; int ti = 0;
            for (int c = 0; c < VT; ++c) {
                float x = Ls[r][c];
                if (x > tm) { tm = x; ti = c; }
            }
            if (tm > amax_v[r]) { amax_v[r] = tm; amax_i[r] = vbase + ti; }
            float mo = m_run[r];
            float mn = fmaxf(mo, tm);
            float s  = s_run[r] * __expf(mo - mn);
            for (int c = 0; c < VT; ++c) s += __expf(Ls[r][c] - mn);
            m_run[r] = mn; s_run[r] = s;
            int lbl = labels[rowBase + r];
            int st  = (lbl == IGNORE) ? 0 : lbl;
            if (st >= vbase && st < vbase + VT) t_run[r] = Ls[r][st - vbase];
        }
        __syncthreads();
    }

    // ---- write per-(row, chunk) partials ----
    if (tid < RPB) {
        long p = (long)(rowBase + tid) * NCHUNK + chunk;
        pm[p]  = m_run[tid];
        ps[p]  = s_run[tid];
        pt[p]  = t_run[tid];
        pav[p] = amax_v[tid];
        pai[p] = amax_i[tid];
    }
}

// -------------------- 3) combine partials -> 3 scalars ---------------------
__global__ __launch_bounds__(256)
void mlm_combine_kernel(const int* __restrict__ labels,
                        const float* __restrict__ pm, const float* __restrict__ ps,
                        const float* __restrict__ pt, const float* __restrict__ pav,
                        const int* __restrict__ pai,
                        float* __restrict__ out) {
    __shared__ float rl[256];
    __shared__ int   rm[256], rc[256];
    float loss = 0.f; int nm = 0, nc = 0;

    for (int row = threadIdx.x; row < NTOK; row += 256) {
        float M = NEG_INF, S = 0.f, T = NEG_INF, AV = NEG_INF;
        int AI = 0;
        for (int c = 0; c < NCHUNK; ++c) {
            long p = (long)row * NCHUNK + c;
            float m = pm[p], s = ps[p];
            float Mn = fmaxf(M, m);
            S = S * __expf(M - Mn) + s * __expf(m - Mn);
            M = Mn;
            T = fmaxf(T, pt[p]);
            float av = pav[p];
            if (av > AV) { AV = av; AI = pai[p]; }   // chunks ascend -> first-max kept
        }
        int lbl = labels[row];
        if (lbl != IGNORE) {
            loss += M + __logf(S) - T;   // -log_softmax[target]
            nm   += 1;
            nc   += (AI == lbl) ? 1 : 0;
        }
    }
    rl[threadIdx.x] = loss; rm[threadIdx.x] = nm; rc[threadIdx.x] = nc;
    __syncthreads();
    for (int off = 128; off > 0; off >>= 1) {
        if (threadIdx.x < (unsigned)off) {
            rl[threadIdx.x] += rl[threadIdx.x + off];
            rm[threadIdx.x] += rm[threadIdx.x + off];
            rc[threadIdx.x] += rc[threadIdx.x + off];
        }
        __syncthreads();
    }
    if (threadIdx.x == 0) {
        out[0] = rl[0];
        out[1] = (float)rm[0];
        out[2] = (float)rc[0];
    }
}

// ---------------------------------------------------------------------------
extern "C" void kernel_launch(void* const* d_in, const int* in_sizes, int n_in,
                              void* d_out, int out_size, void* d_ws, size_t ws_size,
                              hipStream_t stream) {
    const float* Hs     = (const float*)d_in[0];   // (4096, 768)
    const int*   labels = (const int*)d_in[1];     // (4096,)
    const float* W      = (const float*)d_in[2];   // (32000, 768)
    float* out = (float*)d_out;

    // workspace layout: bf16 W | bf16 H | partial stats (m,s,t,amax_v,amax_i)
    char* w = (char*)d_ws;
    __bf16* Wb = (__bf16*)w;
    __bf16* Hb = (__bf16*)(w + (size_t)VOCAB * HDIM * 2);
    float*  pm = (float*)(w + (size_t)VOCAB * HDIM * 2 + (size_t)NTOK * HDIM * 2);
    size_t  P  = (size_t)NTOK * NCHUNK;
    float*  ps  = pm + P;
    float*  pt  = ps + P;
    float*  pav = pt + P;
    int*    pai = (int*)(pav + P);

    long total  = (long)VOCAB * HDIM;               // dominates NTOK*HDIM
    int  blocks = (int)((total + 1023) / 1024);
    convert_bf16_kernel<<<blocks, 256, 0, stream>>>(W, Hs, Wb, Hb);

    dim3 grid(NCHUNK, NROWBLK);                     // 25 x 32 = 800 blocks
    mlm_logits_chunk_kernel<<<grid, 256, 0, stream>>>(Hb, Wb, labels,
                                                      pm, ps, pt, pav, pai);

    mlm_combine_kernel<<<1, 256, 0, stream>>>(labels, pm, ps, pt, pav, pai, out);
}
